// Induction_87196426043660
// MI455X (gfx1250) — compile-verified, matrix operand order checked by reading, not run on hardware
//
#include <hip/hip_runtime.h>
#include <hip/hip_bf16.h>
#include <math.h>

// ---------------------------------------------------------------------------
// Problem: C=512, S=64, H=1024
//   e = (x.reshape(32768,1024) @ W)     -- fp32 GEMM via split-bf16 WMMA
//   2x routing update on b_ij, then final softmax + sum(d_i * x)
// Workspace layout: E (32768*1024 f32 = 128 MiB) | B (512*64 f32 = 128 KiB)
// ---------------------------------------------------------------------------

typedef __attribute__((ext_vector_type(16))) __bf16 v16bf;
typedef __attribute__((ext_vector_type(8)))  __bf16 v8bf;
typedef __attribute__((ext_vector_type(4)))  __bf16 v4bf;
typedef __attribute__((ext_vector_type(8)))  float  v8f;

#define C_CAP 512
#define S_CAP 64
#define H_DIM 1024
#define M_TOT (C_CAP * S_CAP)   // 32768
#define N_TOT H_DIM             // 1024
#define K_TOT H_DIM             // 1024

constexpr int BM  = 128;
constexpr int BN  = 64;
constexpr int BK  = 32;
constexpr int NK  = K_TOT / BK;  // 32
constexpr int LDK = BK + 8;      // 40 bf16 = 80 B rows: 16B-aligned, banks spaced by 4

// ---------------------------------------------------------------------------
// split-bf16 fp32 emulation GEMM: E[m,n] = sum_k X[m,k] * W[k,n]
// f32 -> (hi,lo) bf16 conversion happens ONCE at LDS staging time; the WMMA
// inner loop is pure ds_load_b128 + v_wmma_f32_16x16x32_bf16.
// Per K=32 chunk: acc += ahi*bhi + ahi*blo + alo*bhi  (3 WMMAs, lo*lo dropped)
// ---------------------------------------------------------------------------
__global__ __launch_bounds__(256)
void gemm_bf16split(const float* __restrict__ X, const float* __restrict__ W,
                    float* __restrict__ E)
{
    __shared__ __bf16 sXh[2][BM][LDK];   // X tile, hi part   (row-major [m][k])
    __shared__ __bf16 sXl[2][BM][LDK];   // X tile, lo part
    __shared__ __bf16 sWh[2][BN][LDK];   // W tile TRANSPOSED [n][k], hi part
    __shared__ __bf16 sWl[2][BN][LDK];   // W tile TRANSPOSED [n][k], lo part

    const int tid  = threadIdx.x;
    const int lane = tid & 31;
    const int wave = tid >> 5;       // 0..7
    const int wm   = wave & 3;       // wave row  (4 x 32 rows)
    const int wn   = wave >> 2;      // wave col  (2 x 32 cols)
    const int m0   = blockIdx.y * BM;
    const int n0   = blockIdx.x * BN;

    const int half = lane >> 4;      // which 16-lane half
    const int l16  = lane & 15;
    const int r0   = half * 8;       // frag elems 0..7  -> K = r0..r0+7
    const int r1   = 16 + half * 8;  // frag elems 8..15 -> K = r1..r1+7

    // X staging: 4 float4 per thread (coalesced along K)
    const int xrow = tid >> 3;           // 0..31 (plus i*32)
    const int xc4  = (tid & 7) * 4;      // 0..28
    // W staging: 8 k-strided scalars per thread (coalesced along N per k)
    const int wncol = tid & 63;          // 0..63
    const int wkg   = (tid >> 6) * 8;    // 0,8,16,24

    float4 rx[4];
    float  wf[8];

    auto loadX = [&](int k0) {
#pragma unroll
        for (int i = 0; i < 4; ++i) {
            rx[i] = *reinterpret_cast<const float4*>(
                        &X[(size_t)(m0 + xrow + i * 32) * K_TOT + k0 + xc4]);
        }
    };
    auto loadW = [&](int k0) {
#pragma unroll
        for (int j = 0; j < 8; ++j) {
            wf[j] = W[(size_t)(k0 + wkg + j) * N_TOT + n0 + wncol];
        }
    };
    auto stageX = [&](int buf) {
#pragma unroll
        for (int i = 0; i < 4; ++i) {
            float4 v = rx[i];
            __bf16 h0 = (__bf16)v.x, h1 = (__bf16)v.y,
                   h2 = (__bf16)v.z, h3 = (__bf16)v.w;
            v4bf hi = {h0, h1, h2, h3};
            v4bf lo = {(__bf16)(v.x - (float)h0), (__bf16)(v.y - (float)h1),
                       (__bf16)(v.z - (float)h2), (__bf16)(v.w - (float)h3)};
            *reinterpret_cast<v4bf*>(&sXh[buf][xrow + i * 32][xc4]) = hi;
            *reinterpret_cast<v4bf*>(&sXl[buf][xrow + i * 32][xc4]) = lo;
        }
    };
    auto stageW = [&](int buf) {
        v8bf hi, lo;
#pragma unroll
        for (int j = 0; j < 8; ++j) {
            __bf16 h = (__bf16)wf[j];
            hi[j] = h;
            lo[j] = (__bf16)(wf[j] - (float)h);
        }
        *reinterpret_cast<v8bf*>(&sWh[buf][wncol][wkg]) = hi;
        *reinterpret_cast<v8bf*>(&sWl[buf][wncol][wkg]) = lo;
    };

    union FragU { v16bf v; float4 q[2]; };
    auto loadFrag = [&](const __bf16* rowp) -> v16bf {
        FragU f;
        f.q[0] = *reinterpret_cast<const float4*>(rowp + r0);  // 8 bf16, 16 B
        f.q[1] = *reinterpret_cast<const float4*>(rowp + r1);  // 8 bf16, 16 B
        return f.v;
    };

    v8f acc[2][2] = {};

    loadX(0); loadW(0);
    stageX(0); stageW(0);
    int buf = 0;

    for (int kt = 0; kt < NK; ++kt) {
        __syncthreads();   // stores to `buf` visible; prior reads of `buf^1` done

        if (kt + 1 < NK) {
            loadX((kt + 1) * BK);
            loadW((kt + 1) * BK);
        }
        if (kt + 2 < NK) {  // global_prefetch_b8 for the tile after next
            __builtin_prefetch(&X[(size_t)(m0 + xrow) * K_TOT + (kt + 2) * BK + xc4], 0, 1);
        }

        // Pure-LDS fragment builds (no conversion VALU in the hot path)
        v16bf ah[2], al[2], bh[2], bl[2];
#pragma unroll
        for (int tm = 0; tm < 2; ++tm) {
            const int row = wm * 32 + tm * 16 + l16;
            ah[tm] = loadFrag(&sXh[buf][row][0]);
            al[tm] = loadFrag(&sXl[buf][row][0]);
        }
#pragma unroll
        for (int tn = 0; tn < 2; ++tn) {
            const int row = wn * 32 + tn * 16 + l16;
            bh[tn] = loadFrag(&sWh[buf][row][0]);
            bl[tn] = loadFrag(&sWl[buf][row][0]);
        }

#pragma unroll
        for (int tm = 0; tm < 2; ++tm)
#pragma unroll
            for (int tn = 0; tn < 2; ++tn) {
                acc[tm][tn] = __builtin_amdgcn_wmma_f32_16x16x32_bf16(
                    false, ah[tm], false, bh[tn], (short)0, acc[tm][tn], false, false);
                acc[tm][tn] = __builtin_amdgcn_wmma_f32_16x16x32_bf16(
                    false, ah[tm], false, bl[tn], (short)0, acc[tm][tn], false, false);
                acc[tm][tn] = __builtin_amdgcn_wmma_f32_16x16x32_bf16(
                    false, al[tm], false, bh[tn], (short)0, acc[tm][tn], false, false);
            }

        if (kt + 1 < NK) { stageX(buf ^ 1); stageW(buf ^ 1); buf ^= 1; }
    }

    // C/D layout: VGPR r: lanes 0-15 -> (M = base+r, N = lane), lanes 16-31 -> M = base+8+r
#pragma unroll
    for (int tm = 0; tm < 2; ++tm)
#pragma unroll
        for (int tn = 0; tn < 2; ++tn) {
            int col   = n0 + wn * 32 + tn * 16 + l16;
            int rbase = m0 + wm * 32 + tm * 16 + half * 8;
#pragma unroll
            for (int r = 0; r < 8; ++r)
                E[(size_t)(rbase + r) * N_TOT + col] = acc[tm][tn][r];
        }
}

// ---------------------------------------------------------------------------
// Reductions (wave32)
// ---------------------------------------------------------------------------
__device__ __forceinline__ float waveSum(float v) {
#pragma unroll
    for (int o = 16; o > 0; o >>= 1) v += __shfl_xor(v, o, 32);
    return v;
}
__device__ __forceinline__ float waveMax(float v) {
#pragma unroll
    for (int o = 16; o > 0; o >>= 1) v = fmaxf(v, __shfl_xor(v, o, 32));
    return v;
}

// softmax over 64 entries of Brow into d_s; called by 256-thread block
__device__ __forceinline__ void softmax64(const float* __restrict__ Brow,
                                          float* d_s, float* red,
                                          int tid, int lane, int wave)
{
    float v = (tid < S_CAP) ? Brow[tid] : -3.402823466e38f;
    float m = waveMax(v);
    if (lane == 0) red[wave] = m;
    __syncthreads();
    float bm = fmaxf(fmaxf(fmaxf(red[0], red[1]), fmaxf(red[2], red[3])),
                     fmaxf(fmaxf(red[4], red[5]), fmaxf(red[6], red[7])));
    __syncthreads();
    float ex = (tid < S_CAP) ? __expf(v - bm) : 0.0f;
    float s = waveSum(ex);
    if (lane == 0) red[wave] = s;
    __syncthreads();
    float bs = red[0] + red[1] + red[2] + red[3] + red[4] + red[5] + red[6] + red[7];
    if (tid < S_CAP) d_s[tid] = ex / bs;
    __syncthreads();
}

// ---------------------------------------------------------------------------
// One routing iteration: d = softmax(b); c = norm(sum_s d*e); b = (e @ c)/100
// One block per capsule c. E rows are L2-resident (134 MB < 192 MB L2).
// ---------------------------------------------------------------------------
__global__ __launch_bounds__(256)
void routing_iter(const float* __restrict__ E, float* __restrict__ B)
{
    __shared__ float d_s[S_CAP];
    __shared__ float ci[H_DIM];
    __shared__ float red[8];

    const int c    = blockIdx.x;
    const int tid  = threadIdx.x;
    const int lane = tid & 31;
    const int wave = tid >> 5;

    softmax64(&B[(size_t)c * S_CAP], d_s, red, tid, lane, wave);

    const float* Ec = E + (size_t)c * S_CAP * H_DIM;

    // c_i[h] = sum_s d[s] * e[c,s,h]
    float a[4];
    float sq = 0.0f;
#pragma unroll
    for (int j = 0; j < 4; ++j) {
        int h = tid + j * 256;
        float accv = 0.0f;
        for (int s = 0; s < S_CAP; ++s)
            accv = fmaf(d_s[s], Ec[(size_t)s * H_DIM + h], accv);
        a[j] = accv;
        sq += accv * accv;
    }
    float t = waveSum(sq);
    if (lane == 0) red[wave] = t;
    __syncthreads();
    float tot = red[0] + red[1] + red[2] + red[3] + red[4] + red[5] + red[6] + red[7];
    float inv = rsqrtf(tot + 1e-9f);
#pragma unroll
    for (int j = 0; j < 4; ++j) ci[tid + j * 256] = a[j] * inv;
    __syncthreads();

    // b[c,s] = dot(e[c,s,:], c_i) / 100 ; wave w handles s = w, w+8, ...
    for (int i = 0; i < 8; ++i) {
        int s = wave + 8 * i;
        const float* row = Ec + (size_t)s * H_DIM;
        float p = 0.0f;
#pragma unroll
        for (int j = 0; j < 32; ++j)
            p = fmaf(row[lane + 32 * j], ci[lane + 32 * j], p);
        p = waveSum(p);
        if (lane == 0) B[(size_t)c * S_CAP + s] = p / 100.0f;
    }
}

// ---------------------------------------------------------------------------
// Final: d = softmax(b); out[c,h] = sum_s d[s] * x[c,s,h]
// ---------------------------------------------------------------------------
__global__ __launch_bounds__(256)
void routing_final(const float* __restrict__ X, const float* __restrict__ B,
                   float* __restrict__ Out)
{
    __shared__ float d_s[S_CAP];
    __shared__ float red[8];

    const int c    = blockIdx.x;
    const int tid  = threadIdx.x;
    const int lane = tid & 31;
    const int wave = tid >> 5;

    softmax64(&B[(size_t)c * S_CAP], d_s, red, tid, lane, wave);

    const float* Xc = X + (size_t)c * S_CAP * H_DIM;
#pragma unroll
    for (int j = 0; j < 4; ++j) {
        int h = tid + j * 256;
        float accv = 0.0f;
        for (int s = 0; s < S_CAP; ++s)
            accv = fmaf(d_s[s], Xc[(size_t)s * H_DIM + h], accv);
        Out[(size_t)c * H_DIM + h] = accv;
    }
}

__global__ void zero_b(float* __restrict__ B)
{
    B[blockIdx.x * 256 + threadIdx.x] = 0.0f;
}

// ---------------------------------------------------------------------------
extern "C" void kernel_launch(void* const* d_in, const int* in_sizes, int n_in,
                              void* d_out, int out_size, void* d_ws, size_t ws_size,
                              hipStream_t stream)
{
    (void)in_sizes; (void)n_in; (void)out_size; (void)ws_size;

    const float* x = (const float*)d_in[0];   // [512, 64, 1024] fp32
    const float* W = (const float*)d_in[1];   // [1024, 1024] fp32
    float* out = (float*)d_out;               // [512, 1024] fp32

    float* E = (float*)d_ws;                                           // 128 MiB
    float* B = (float*)((char*)d_ws + (size_t)M_TOT * N_TOT * sizeof(float));

    // b_ij starts at zero (workspace is poisoned by harness)
    zero_b<<<M_TOT / 256, 256, 0, stream>>>(B);

    dim3 grid(N_TOT / BN, M_TOT / BM);        // (16, 256)
    gemm_bf16split<<<grid, 256, 0, stream>>>(x, W, E);

    // iterations 1 and 2 update b_ij; iteration 3's c_i/b_ij are dead code
    routing_iter<<<C_CAP, 256, 0, stream>>>(E, B);
    routing_iter<<<C_CAP, 256, 0, stream>>>(E, B);

    // iteration 3: softmax of final b_ij, applied to the ORIGINAL x
    routing_final<<<C_CAP, 256, 0, stream>>>(x, B, out);
}
